// CrossAttentionBlock_84061099917430
// MI455X (gfx1250) — matrix operand order repaired and checked by reading, not compiled, not run on hardware
//
#include <hip/hip_runtime.h>
#include <hip/hip_bf16.h>
#include <math.h>

// ---------------------------------------------------------------------------
// CrossAttentionBlock for MI455X (gfx1250, wave32, WMMA).
// All GEMMs run on v_wmma_f32_16x16x32_f16 (f16 in, f32 accum).
// ---------------------------------------------------------------------------

typedef __attribute__((ext_vector_type(16))) _Float16 v16h;
typedef __attribute__((ext_vector_type(8)))  float    v8f;

#define BATCH      16
#define HIDDEN     512
#define TEXT_DIM   768
#define NUM_HEADS  8
#define HEAD_DIM   64
#define NUM_GROUPS 32
#define N_IMG      4096      // 64*64
#define N_TEXT     77
#define N_TEXT_R16 80        // 77 padded to 5 x 16 rows (kv GEMM)
#define N_TEXT_PAD 96        // 3 x K32 chunks (attention K-dim)

__device__ __forceinline__ v8f wmma_f16(v16h a, v16h b, v8f c) {
  // 8 args: (neg_a, A, neg_b, B, c_mod, C, reuse_a, reuse_b)
  return __builtin_amdgcn_wmma_f32_16x16x32_f16(false, a, false, b, (short)0, c,
                                                false, false);
}

// Load 16 halfs as two contiguous 16-byte chunks (A/B operand fragments).
__device__ __forceinline__ v16h ld16h(const _Float16* plo, const _Float16* phi) {
  union { v16h v; uint4 q[2]; } u;
  u.q[0] = *(const uint4*)plo;
  u.q[1] = *(const uint4*)phi;
  return u.v;
}

__device__ __forceinline__ v8f zero8f() {
  v8f z = {0.f, 0.f, 0.f, 0.f, 0.f, 0.f, 0.f, 0.f};
  return z;
}

// ---------------------------------------------------------------------------
// Elementwise helpers
// ---------------------------------------------------------------------------
__global__ void f32_to_f16_kernel(const float* __restrict__ src,
                                  _Float16* __restrict__ dst, int n) {
  int i = blockIdx.x * blockDim.x + threadIdx.x;
  if (i < n) dst[i] = (_Float16)src[i];
}

__global__ void zero_f16_kernel(_Float16* __restrict__ dst, int n) {
  int i = blockIdx.x * blockDim.x + threadIdx.x;
  if (i < n) dst[i] = (_Float16)0.f;
}

// text [B,77,768] f32 -> textH [B,80,768] f16, rows 77..79 zero.
__global__ void text_pad_kernel(const float* __restrict__ text,
                                _Float16* __restrict__ textH) {
  int idx = blockIdx.x * blockDim.x + threadIdx.x;   // total = 16*80*768
  int c = idx % TEXT_DIM;
  int t = (idx / TEXT_DIM) % N_TEXT_R16;
  int b = idx / (TEXT_DIM * N_TEXT_R16);
  _Float16 v = (_Float16)0.f;
  if (t < N_TEXT) v = (_Float16)text[((size_t)(b * N_TEXT + t)) * TEXT_DIM + c];
  textH[idx] = v;
}

// ---------------------------------------------------------------------------
// GroupNorm: stats per (b, group) over 16 channels * 4096 pixels (contiguous).
// ---------------------------------------------------------------------------
__global__ void gn_stats_kernel(const float* __restrict__ x,
                                float* __restrict__ stats) {
  __shared__ float ssum[256], ssq[256];
  int bg = blockIdx.x;                                  // b*32 + g
  const float* base = x + (size_t)bg * (16 * N_IMG);    // group is contiguous
  float s = 0.f, sq = 0.f;
  for (int i = threadIdx.x; i < 16 * N_IMG; i += 256) {
    float v = base[i];
    s += v; sq += v * v;
  }
  ssum[threadIdx.x] = s; ssq[threadIdx.x] = sq;
  __syncthreads();
  for (int o = 128; o > 0; o >>= 1) {
    if (threadIdx.x < o) {
      ssum[threadIdx.x] += ssum[threadIdx.x + o];
      ssq[threadIdx.x]  += ssq[threadIdx.x + o];
    }
    __syncthreads();
  }
  if (threadIdx.x == 0) {
    const float inv_n = 1.f / (16.f * (float)N_IMG);
    float mean = ssum[0] * inv_n;
    float var  = ssq[0] * inv_n - mean * mean;
    stats[bg * 2 + 0] = mean;
    stats[bg * 2 + 1] = rsqrtf(var + 1e-5f);
  }
}

// Normalize + affine + LDS-tiled transpose [B,C,N] f32 -> [B,N,C] f16.
// 64x64 tile; global loads are 32-bit coalesced, stores 128-bit coalesced.
// grid = (N_IMG/64, HIDDEN/64, B), block = 256.
__global__ __launch_bounds__(256)
void gn_apply_kernel(const float* __restrict__ x,
                     const float* __restrict__ stats,
                     const float* __restrict__ w,
                     const float* __restrict__ bias,
                     _Float16* __restrict__ xh) {
  __shared__ _Float16 tile[64][72];     // [nn][cc], stride padded vs bank conflicts
  int n0 = blockIdx.x * 64;
  int c0 = blockIdx.y * 64;
  int b  = blockIdx.z;

  int nn = threadIdx.x & 63;
  int cq = threadIdx.x >> 6;            // 0..3
#pragma unroll 4
  for (int pass = 0; pass < 16; pass++) {
    int cc = pass * 4 + cq;
    int c  = c0 + cc;
    int g  = c >> 4;
    float mean = stats[(b * 32 + g) * 2 + 0];
    float rstd = stats[(b * 32 + g) * 2 + 1];
    float v = x[((size_t)(b * HIDDEN + c)) * N_IMG + n0 + nn];
    tile[nn][cc] = (_Float16)((v - mean) * rstd * w[c] + bias[c]);
  }
  __syncthreads();

  // Drain: 8 consecutive channels per thread -> 16B stores, n-major coalesced.
#pragma unroll
  for (int it = 0; it < 2; it++) {
    int seg = it * 256 + threadIdx.x;   // 0..511
    int row = seg >> 3;                 // nn 0..63
    int cc8 = (seg & 7) * 8;
    uint4 q = *(const uint4*)&tile[row][cc8];
    *(uint4*)&xh[((size_t)(b * N_IMG + n0 + row)) * HIDDEN + c0 + cc8] = q;
  }
}

// ---------------------------------------------------------------------------
// K/V projections as WMMA GEMM + fused per-head L2 norm of k.
// grid = (B, nh); block = 320 (10 waves): waves 0..4 -> K row-tiles,
// waves 5..9 -> V row-tiles (80 padded text rows, K-dim 768 = 24 steps).
// kh: [B, nh, 96, 64] f16 (t>=80 pre-zeroed; 77..79 come out zero).
// vT: [B, nh, 64, 96] f16 (t>=80 pre-zeroed).
// ---------------------------------------------------------------------------
__global__ __launch_bounds__(320)
void kv_wmma_kernel(const _Float16* __restrict__ textH,   // [B*80][768]
                    const _Float16* __restrict__ WkH,     // [512][768]
                    const _Float16* __restrict__ WvH,     // [512][768]
                    _Float16* __restrict__ kh,
                    _Float16* __restrict__ vT) {
  int lane = threadIdx.x & 31;
  int wave = threadIdx.x >> 5;
  int m = lane & 15, hs = lane >> 4;
  int b = blockIdx.x, h = blockIdx.y;
  bool isV = wave >= 5;
  int tile = isV ? wave - 5 : wave;
  int t0 = tile * 16;
  const _Float16* W = isV ? WvH : WkH;
  int cbase = h * 64;

  v8f acc[4] = {zero8f(), zero8f(), zero8f(), zero8f()};
  for (int k0 = 0; k0 < TEXT_DIM; k0 += 32) {
    const _Float16* arow = textH + ((size_t)(b * N_TEXT_R16 + t0 + m)) * TEXT_DIM + k0;
    v16h a = ld16h(arow + hs * 8, arow + 16 + hs * 8);
#pragma unroll
    for (int j = 0; j < 4; j++) {
      const _Float16* brow = W + ((size_t)(cbase + j * 16 + m)) * TEXT_DIM + k0;
      v16h bb = ld16h(brow + hs * 8, brow + 16 + hs * 8);
      acc[j] = wmma_f16(a, bb, acc[j]);
    }
  }

  size_t bh = (size_t)b * NUM_HEADS + h;
  if (!isV) {
#pragma unroll
    for (int r = 0; r < 8; r++) {
      float s = acc[0][r] * acc[0][r] + acc[1][r] * acc[1][r] +
                acc[2][r] * acc[2][r] + acc[3][r] * acc[3][r];
#pragma unroll
      for (int off = 1; off < 16; off <<= 1) s += __shfl_xor(s, off, 32);
      float rn = 1.f / (sqrtf(s) + 1e-6f);
      int t = t0 + r + hs * 8;
      _Float16* dst = kh + (bh * N_TEXT_PAD + t) * HEAD_DIM;
#pragma unroll
      for (int j = 0; j < 4; j++) dst[j * 16 + m] = (_Float16)(acc[j][r] * rn);
    }
  } else {
#pragma unroll
    for (int r = 0; r < 8; r++) {
      int t = t0 + r + hs * 8;
#pragma unroll
      for (int j = 0; j < 4; j++) {
        int d = j * 16 + m;
        vT[(bh * HEAD_DIM + d) * N_TEXT_PAD + t] = (_Float16)acc[j][r];
      }
    }
  }
}

// ---------------------------------------------------------------------------
// 512-wide projection GEMM, WMMA f16.  Y[row, col] = X[row,:] . W[col,:]
// Block = 256 (8 waves); wave owns 32 rows x 64 cols (acc[2][4]);
// grid = (B*N_IMG/256, 8).
// MODE 0 (Q): blockIdx.y == head; fused per-query L2 norm; f16 out [B,nh,N,64].
// MODE 1 (O): + bias + residual via LDS-staged coalesced epilogue; fp32 BCHW.
// ---------------------------------------------------------------------------
template <int MODE>
__global__ __launch_bounds__(256)
void proj_wmma_kernel(const _Float16* __restrict__ X,   // [B*N_IMG][512]
                      const _Float16* __restrict__ W,   // [512][512] (rows = out cols)
                      _Float16* __restrict__ qh,        // MODE 0
                      const float* __restrict__ bo,     // MODE 1
                      const float* __restrict__ xres,   // MODE 1, [B,C,N]
                      float* __restrict__ out) {        // MODE 1
  __shared__ float otile[32 * 257];     // MODE 1 epilogue staging (pad stride)
  int lane = threadIdx.x & 31;
  int wave = threadIdx.x >> 5;
  int m = lane & 15, hs = lane >> 4;
  int rbase = blockIdx.x * 256 + wave * 32;   // 4096 % 256 == 0: no batch crossing
  int cbase = blockIdx.y * 64;

  v8f acc[2][4] = {{zero8f(), zero8f(), zero8f(), zero8f()},
                   {zero8f(), zero8f(), zero8f(), zero8f()}};

  for (int k0 = 0; k0 < HIDDEN; k0 += 32) {
    v16h a[2];
#pragma unroll
    for (int i = 0; i < 2; i++) {
      const _Float16* arow = X + ((size_t)(rbase + i * 16 + m)) * HIDDEN + k0;
      a[i] = ld16h(arow + hs * 8, arow + 16 + hs * 8);
    }
#pragma unroll
    for (int j = 0; j < 4; j++) {
      const _Float16* brow = W + ((size_t)(cbase + j * 16 + m)) * HIDDEN + k0;
      v16h bb = ld16h(brow + hs * 8, brow + 16 + hs * 8);
#pragma unroll
      for (int i = 0; i < 2; i++) acc[i][j] = wmma_f16(a[i], bb, acc[i][j]);
    }
  }

  if (MODE == 0) {
    int b  = rbase >> 12;
    int n0 = rbase & 4095;
    int h = blockIdx.y;       // 64-col tile == exactly one head
#pragma unroll
    for (int i = 0; i < 2; i++) {
#pragma unroll
      for (int r = 0; r < 8; r++) {
        // C-layout: vgpr r holds row (r + 8*hs), col = lane&15 of tile j.
        float s = acc[i][0][r] * acc[i][0][r] + acc[i][1][r] * acc[i][1][r] +
                  acc[i][2][r] * acc[i][2][r] + acc[i][3][r] * acc[i][3][r];
#pragma unroll
        for (int off = 1; off < 16; off <<= 1) s += __shfl_xor(s, off, 32);
        float rn = 1.f / (sqrtf(s) + 1e-6f);
        int row = n0 + i * 16 + r + hs * 8;
        _Float16* dst = qh + (((size_t)(b * NUM_HEADS + h)) * N_IMG + row) * HEAD_DIM;
#pragma unroll
        for (int j = 0; j < 4; j++) dst[j * 16 + m] = (_Float16)(acc[i][j][r] * rn);
      }
    }
  } else {
    int rb0 = blockIdx.x * 256;
    int b   = rb0 >> 12;
    int n0  = rb0 & 4095;
    // Two passes of 32 cols: stage in LDS, then drain n-contiguous (coalesced
    // residual reads + output writes in [B,C,N] order).
#pragma unroll
    for (int p = 0; p < 2; p++) {
      __syncthreads();
#pragma unroll
      for (int i = 0; i < 2; i++) {
#pragma unroll
        for (int jj = 0; jj < 2; jj++) {
          int j = p * 2 + jj;
          int col_l = jj * 16 + m;
          int row_l = wave * 32 + i * 16 + hs * 8;
#pragma unroll
          for (int r = 0; r < 8; r++)
            otile[col_l * 257 + row_l + r] = acc[i][j][r];
        }
      }
      __syncthreads();
      for (int idx = threadIdx.x; idx < 32 * 256; idx += 256) {
        int n   = idx & 255;
        int col = idx >> 8;
        int gcol = cbase + p * 32 + col;
        size_t oi = ((size_t)(b * HIDDEN + gcol)) * N_IMG + n0 + n;
        out[oi] = xres[oi] + otile[col * 257 + n] + bo[gcol];
      }
    }
  }
}

// ---------------------------------------------------------------------------
// Fused cosine-sim attention. 128 threads = 4 waves; wave owns 16 queries.
// scores (12 WMMA) -> LDS -> masked softmax over 77 keys -> attn@V (12 WMMA).
// grid = B * nh * (4096/64) = 8192.
// ---------------------------------------------------------------------------
__global__ __launch_bounds__(128)
void attn_kernel(const _Float16* __restrict__ qh,     // [B,nh,4096,64]
                 const _Float16* __restrict__ kh,     // [B,nh,96,64]
                 const _Float16* __restrict__ vT,     // [B,nh,64,96]
                 _Float16* __restrict__ attout) {     // [B,4096,512]
  __shared__ __align__(16) float    sc[4][16][N_TEXT_PAD];
  __shared__ __align__(16) _Float16 at[4][16][N_TEXT_PAD];

  int lane = threadIdx.x & 31;
  int wave = threadIdx.x >> 5;
  int m = lane & 15, hs = lane >> 4;
  int bh = blockIdx.x >> 6;
  int n0 = (blockIdx.x & 63) * 64 + wave * 16;
  int b = bh >> 3, h = bh & 7;

  // Q fragment: A operand for K=0..31 and K=32..63 (q already L2-normalized).
  const _Float16* qrow = qh + ((size_t)bh * N_IMG + n0 + m) * HEAD_DIM;
  v16h a0 = ld16h(qrow + hs * 8, qrow + 16 + hs * 8);
  v16h a1 = ld16h(qrow + 32 + hs * 8, qrow + 48 + hs * 8);

  // scores[16 x 96] = q(16x64) x k^T(64x96), 6 column tiles of 16 keys.
#pragma unroll
  for (int j = 0; j < 6; j++) {
    const _Float16* kr = kh + ((size_t)bh * N_TEXT_PAD + j * 16 + m) * HEAD_DIM;
    v16h b0 = ld16h(kr + hs * 8, kr + 16 + hs * 8);             // d = 0..31
    v16h b1 = ld16h(kr + 32 + hs * 8, kr + 48 + hs * 8);        // d = 32..63
    v8f s = zero8f();
    s = wmma_f16(a0, b0, s);
    s = wmma_f16(a1, b1, s);
#pragma unroll
    for (int r = 0; r < 8; r++) sc[wave][r + hs * 8][j * 16 + m] = s[r];
  }
  __syncthreads();

  // Softmax over the 77 real keys; one lane per query row.
  if (lane < 16) {
    float* row = sc[wave][lane];
    float mx = -1e30f;
    for (int t = 0; t < N_TEXT; t++) mx = fmaxf(mx, row[t]);
    float sum = 0.f;
    for (int t = 0; t < N_TEXT; t++) {
      float e = __expf(row[t] - mx);
      row[t] = e;
      sum += e;
    }
    float inv = 1.f / sum;
    _Float16* arow = at[wave][lane];
    for (int t = 0; t < N_TEXT; t++)          arow[t] = (_Float16)(row[t] * inv);
    for (int t = N_TEXT; t < N_TEXT_PAD; t++) arow[t] = (_Float16)0.f;
  }
  __syncthreads();

  // out(16x64) = attn(16x96) x v(96x64): 3 K-chunks x 4 column tiles.
  v8f acc[4] = {zero8f(), zero8f(), zero8f(), zero8f()};
#pragma unroll
  for (int c = 0; c < 3; c++) {
    const _Float16* ar = &at[wave][m][c * 32];
    v16h a = ld16h(ar + hs * 8, ar + 16 + hs * 8);
#pragma unroll
    for (int j = 0; j < 4; j++) {
      const _Float16* vr =
          vT + ((size_t)bh * HEAD_DIM + j * 16 + m) * N_TEXT_PAD + c * 32;
      v16h bb = ld16h(vr + hs * 8, vr + 16 + hs * 8);
      acc[j] = wmma_f16(a, bb, acc[j]);
    }
  }
#pragma unroll
  for (int r = 0; r < 8; r++) {
    int n = n0 + r + hs * 8;
    _Float16* dst = attout + ((size_t)b * N_IMG + n) * HIDDEN + h * HEAD_DIM;
#pragma unroll
    for (int j = 0; j < 4; j++) dst[j * 16 + m] = (_Float16)acc[j][r];
  }
}

// ---------------------------------------------------------------------------
// Host-side launch
// ---------------------------------------------------------------------------
extern "C" void kernel_launch(void* const* d_in, const int* in_sizes, int n_in,
                              void* d_out, int out_size, void* d_ws, size_t ws_size,
                              hipStream_t stream) {
  (void)in_sizes; (void)n_in; (void)out_size; (void)ws_size;
  const float* x    = (const float*)d_in[0];
  const float* text = (const float*)d_in[1];
  const float* gnw  = (const float*)d_in[2];
  const float* gnb  = (const float*)d_in[3];
  const float* Wq   = (const float*)d_in[4];
  const float* Wk   = (const float*)d_in[5];
  const float* Wv   = (const float*)d_in[6];
  const float* Wo   = (const float*)d_in[7];
  const float* bo   = (const float*)d_in[8];
  float* out = (float*)d_out;

  char* p = (char*)d_ws;
  auto alloc = [&](size_t bytes) {
    char* r = p;
    p += (bytes + 255) & ~(size_t)255;
    return r;
  };
  _Float16* xh    = (_Float16*)alloc((size_t)BATCH * N_IMG * HIDDEN * 2);        // 64 MiB
  _Float16* qhb   = (_Float16*)alloc((size_t)BATCH * NUM_HEADS * N_IMG * HEAD_DIM * 2);
  _Float16* ao    = (_Float16*)alloc((size_t)BATCH * N_IMG * HIDDEN * 2);
  _Float16* khb   = (_Float16*)alloc((size_t)BATCH * NUM_HEADS * N_TEXT_PAD * HEAD_DIM * 2);
  _Float16* vTb   = (_Float16*)alloc((size_t)BATCH * NUM_HEADS * HEAD_DIM * N_TEXT_PAD * 2);
  _Float16* WqH   = (_Float16*)alloc((size_t)HIDDEN * HIDDEN * 2);
  _Float16* WoH   = (_Float16*)alloc((size_t)HIDDEN * HIDDEN * 2);
  _Float16* WkH   = (_Float16*)alloc((size_t)HIDDEN * TEXT_DIM * 2);
  _Float16* WvH   = (_Float16*)alloc((size_t)HIDDEN * TEXT_DIM * 2);
  _Float16* textH = (_Float16*)alloc((size_t)BATCH * N_TEXT_R16 * TEXT_DIM * 2);
  float*    stats = (float*)alloc((size_t)BATCH * NUM_GROUPS * 2 * 4);

  const int nW  = HIDDEN * HIDDEN;
  const int nWt = HIDDEN * TEXT_DIM;
  const int nKV = BATCH * NUM_HEADS * N_TEXT_PAD * HEAD_DIM;
  const int nTx = BATCH * N_TEXT_R16 * TEXT_DIM;

  f32_to_f16_kernel<<<(nW + 255) / 256, 256, 0, stream>>>(Wq, WqH, nW);
  f32_to_f16_kernel<<<(nW + 255) / 256, 256, 0, stream>>>(Wo, WoH, nW);
  f32_to_f16_kernel<<<(nWt + 255) / 256, 256, 0, stream>>>(Wk, WkH, nWt);
  f32_to_f16_kernel<<<(nWt + 255) / 256, 256, 0, stream>>>(Wv, WvH, nWt);
  zero_f16_kernel<<<(nKV + 255) / 256, 256, 0, stream>>>(khb, nKV);
  zero_f16_kernel<<<(nKV + 255) / 256, 256, 0, stream>>>(vTb, nKV);
  text_pad_kernel<<<(nTx + 255) / 256, 256, 0, stream>>>(text, textH);

  gn_stats_kernel<<<BATCH * NUM_GROUPS, 256, 0, stream>>>(x, stats);
  gn_apply_kernel<<<dim3(N_IMG / 64, HIDDEN / 64, BATCH), 256, 0, stream>>>(
      x, stats, gnw, gnb, xh);

  kv_wmma_kernel<<<dim3(BATCH, NUM_HEADS), 320, 0, stream>>>(textH, WkH, WvH, khb, vTb);

  proj_wmma_kernel<0><<<dim3((BATCH * N_IMG) / 256, NUM_HEADS), 256, 0, stream>>>(
      xh, WqH, qhb, nullptr, nullptr, nullptr);

  attn_kernel<<<BATCH * NUM_HEADS * (N_IMG / 64), 128, 0, stream>>>(qhb, khb, vTb, ao);

  proj_wmma_kernel<1><<<dim3((BATCH * N_IMG) / 256, NUM_HEADS), 256, 0, stream>>>(
      ao, WoH, nullptr, bo, x, out);
}
